// GraphSAGE_1666447311245
// MI455X (gfx1250) — matrix operand order repaired and checked by reading, not compiled, not run on hardware
//
#include <hip/hip_runtime.h>
#include <math.h>

typedef __attribute__((ext_vector_type(16))) _Float16 v16h;
typedef __attribute__((ext_vector_type(8)))  _Float16 v8h;
typedef __attribute__((ext_vector_type(8)))  float    v8f;

#define DH    128
#define DOUT  10
#define NGR   64

// ---------------- degree / normalization ----------------
__global__ void k_deg_init(float* deg, int n) {
  int i = blockIdx.x * blockDim.x + threadIdx.x;
  if (i < n) deg[i] = 1.0f;                       // self loop contributes 1
}
__global__ void k_deg_accum(const int* __restrict__ dst, float* deg, int e) {
  int i = blockIdx.x * blockDim.x + threadIdx.x;
  if (i < e) atomicAdd(&deg[dst[i]], 1.0f);
}
__global__ void k_deg_rsqrt(float* deg, int n) {
  int i = blockIdx.x * blockDim.x + threadIdx.x;
  if (i < n) deg[i] = rsqrtf(deg[i]);             // deg>0 always (self loop)
}

// ---------------- precision staging ----------------
// fp32 -> f16 elementwise (for activations / input x)
__global__ void k_cvt_h(const float* __restrict__ s, _Float16* __restrict__ d, int count) {
  int t = blockIdx.x * blockDim.x + threadIdx.x;
  if (t < count) d[t] = (_Float16)s[t];
}
// W[k][c] fp32 -> Bt[c][k] f16 (transpose so B-fragment K-values are contiguous)
__global__ void k_cvt_wT(const float* __restrict__ W, _Float16* __restrict__ Bt) {
  int t = blockIdx.x * blockDim.x + threadIdx.x;   // 128*128 threads
  int k = t >> 7, c = t & (DH - 1);
  Bt[(size_t)c * DH + k] = (_Float16)W[(size_t)k * DH + c];
}
// relu in place + f16 copy for the next GEMM
__global__ void k_relu_cvt(float* __restrict__ a, _Float16* __restrict__ ah, int count) {
  int t = blockIdx.x * blockDim.x + threadIdx.x;
  if (t < count) {
    float v = fmaxf(a[t], 0.0f);
    a[t] = v;
    ah[t] = (_Float16)v;
  }
}

// ---------------- WMMA GEMM: H[n x 128] = A[n x 128] * W[128 x 128] ----------
// A: row-major f16. Bt: f16, Bt[col][k] = W[k][col]. 256-thread block = 8
// waves; block owns a 16-row tile, wave w owns column tile w. K = 4 x 32.
__global__ void k_gemm128h(const _Float16* __restrict__ A,
                           const _Float16* __restrict__ Bt,
                           float* __restrict__ H, int n) {
  const int wave = threadIdx.x >> 5;
  const int lane = threadIdx.x & 31;
  const int hi   = lane >> 4;          // lane-half: 0 or 1
  const int l    = lane & 15;
  const int rowBase = blockIdx.x * 16;
  const int colBase = wave * 16;

  // clamp (not predicate) so all loads stay unconditional & vectorized;
  // out-of-range rows compute garbage that the store path drops.
  const int arow = min(rowBase + l, n - 1);
  const _Float16* ap = A + (size_t)arow * DH;           // this lane's A row
  const _Float16* bp = Bt + (size_t)(colBase + l) * DH; // this lane's B column

  v8f acc = {};
  #pragma unroll
  for (int k0 = 0; k0 < DH; k0 += 32) {
    // A 16x32 frag: lanes 0-15 hold K=k0+[0..7],[16..23]; lanes 16-31 +8
    v8h a0 = *(const v8h*)(ap + k0 + hi * 8);
    v8h a1 = *(const v8h*)(ap + k0 + hi * 8 + 16);
    v16h a = __builtin_shufflevector(a0, a1, 0,1,2,3,4,5,6,7,8,9,10,11,12,13,14,15);
    // B 32x16 frag: lane = column; lanes 0-15 K=k0+[0..7],[16..23]; lanes 16-31 +8
    v8h b0 = *(const v8h*)(bp + k0 + hi * 8);
    v8h b1 = *(const v8h*)(bp + k0 + hi * 8 + 16);
    v16h b = __builtin_shufflevector(b0, b1, 0,1,2,3,4,5,6,7,8,9,10,11,12,13,14,15);

    acc = __builtin_amdgcn_wmma_f32_16x16x32_f16(
        /*neg_a=*/false, a, /*neg_b=*/false, b,
        /*c_mod=*/(short)0, acc, /*reuse_a=*/false, /*reuse_b=*/false);
  }

  // D store: VGPR r -> row M=r (lanes 0-15) / M=8+r (lanes 16-31), col=l.
  // Full-tile test is wave-uniform -> single scalar branch; fast path is 8
  // unconditional b32 stores at immediate offsets from one base address.
  float* op = H + ((size_t)(rowBase + hi * 8)) * DH + colBase + l;
  if (rowBase + 16 <= n) {
    #pragma unroll
    for (int r = 0; r < 8; ++r) op[(size_t)r * DH] = acc[r];
  } else {
    #pragma unroll
    for (int r = 0; r < 8; ++r) {
      if (rowBase + hi * 8 + r < n) op[(size_t)r * DH] = acc[r];
    }
  }
}

// ---------------- self-loop + bias init: out = dinv[i]^2 * h + b ------------
__global__ void k_agg_init(const float* __restrict__ H, const float* __restrict__ dinv,
                           const float* __restrict__ bias, float* __restrict__ out, int n) {
  int t = blockIdx.x * blockDim.x + threadIdx.x;
  if (t >= n * DH) return;
  int node = t >> 7;
  int c = t & (DH - 1);
  float di = dinv[node];
  out[t] = di * di * H[t] + bias[c];
}

// ---------------- edge scatter: out[dst] += dinv[s]*dinv[d]*H[src] ----------
// One wave per edge, 4 floats per lane (128 features).
__global__ void k_scatter(const float* __restrict__ H, const float* __restrict__ dinv,
                          const int* __restrict__ src, const int* __restrict__ dst,
                          float* __restrict__ out, int e) {
  int t = blockIdx.x * blockDim.x + threadIdx.x;
  int edge = t >> 5;
  if (edge >= e) return;
  int c = (t & 31) * 4;
  int s = src[edge], d = dst[edge];
  float norm = dinv[s] * dinv[d];
  const float4 h4 = *(const float4*)(H + (size_t)s * DH + c);
  float* od = out + (size_t)d * DH + c;
  atomicAdd(od + 0, norm * h4.x);
  atomicAdd(od + 1, norm * h4.y);
  atomicAdd(od + 2, norm * h4.z);
  atomicAdd(od + 3, norm * h4.w);
}

// ---------------- global mean pool + linear head + log_softmax --------------
__global__ void k_pool_zero(float* pooled, float* counts) {
  int t = blockIdx.x * blockDim.x + threadIdx.x;
  if (t < NGR * DH) pooled[t] = 0.0f;
  if (t < NGR) counts[t] = 0.0f;
}
__global__ void k_pool_accum(const float* __restrict__ H, const int* __restrict__ batch,
                             float* pooled, float* counts, int n) {
  int t = blockIdx.x * blockDim.x + threadIdx.x;
  if (t >= n * DH) return;
  int node = t >> 7;
  int c = t & (DH - 1);
  int g = batch[node];
  atomicAdd(&pooled[g * DH + c], H[t]);
  if (c == 0) atomicAdd(&counts[g], 1.0f);
}
__global__ void k_head(const float* __restrict__ pooled, const float* __restrict__ counts,
                       const float* __restrict__ Wl, const float* __restrict__ bl,
                       float* __restrict__ out) {
  __shared__ float p[DH];
  __shared__ float logits[DOUT];
  int g = blockIdx.x;
  int t = threadIdx.x;
  float cnt = fmaxf(counts[g], 1.0f);
  p[t] = pooled[g * DH + t] / cnt;
  __syncthreads();
  if (t < DOUT) {
    float acc = bl[t];
    #pragma unroll 8
    for (int k = 0; k < DH; ++k) acc += p[k] * Wl[k * DOUT + t];
    logits[t] = acc;
  }
  __syncthreads();
  if (t == 0) {
    float m = logits[0];
    for (int i = 1; i < DOUT; ++i) m = fmaxf(m, logits[i]);
    float se = 0.0f;
    for (int i = 0; i < DOUT; ++i) se += expf(logits[i] - m);
    float lse = m + logf(se);
    for (int i = 0; i < DOUT; ++i) out[g * DOUT + i] = logits[i] - lse;
  }
}

extern "C" void kernel_launch(void* const* d_in, const int* in_sizes, int n_in,
                              void* d_out, int out_size, void* d_ws, size_t ws_size,
                              hipStream_t stream) {
  const float* x   = (const float*)d_in[0];
  const int*   src = (const int*)d_in[1];
  const int*   dst = (const int*)d_in[2];
  const int*   bat = (const int*)d_in[3];
  const float* W1  = (const float*)d_in[4];
  const float* b1  = (const float*)d_in[5];
  const float* W2  = (const float*)d_in[6];
  const float* b2  = (const float*)d_in[7];
  const float* W3  = (const float*)d_in[8];
  const float* b3  = (const float*)d_in[9];
  const float* Wl  = (const float*)d_in[10];
  const float* bl  = (const float*)d_in[11];
  float* out = (float*)d_out;

  const int n = in_sizes[3];   // nodes (batch array length)
  const int e = in_sizes[1];   // edges
  const int nd = n * DH;

  // workspace layout (float-sized units, all 16B-aligned slices)
  float* ws = (float*)d_ws;
  size_t nPad = ((size_t)n + 63) & ~(size_t)63;
  float*    deg    = ws;                            // n (becomes dinv in place)
  float*    hbuf   = deg + nPad;                    // n*128 f32 (GEMM out)
  float*    act    = hbuf + (size_t)nd;             // n*128 f32 (activation)
  float*    pooled = act + (size_t)nd;              // 64*128
  float*    counts = pooled + NGR * DH;             // 64
  _Float16* actH   = (_Float16*)(counts + 64);      // n*128 f16 (GEMM A input)
  _Float16* wh     = (_Float16*)((float*)actH + (size_t)nd / 2); // 128*128 f16

  const dim3 blk(256);
  const int gN  = (n + 255) / 256;
  const int gE  = (e + 255) / 256;
  const int gND = (nd + 255) / 256;
  const int gED = ((e * 32) + 255) / 256;
  const int gW  = (DH * DH + 255) / 256;
  const int mTiles = (n + 15) / 16;

  // degree -> dinv
  k_deg_init <<<gN, blk, 0, stream>>>(deg, n);
  k_deg_accum<<<gE, blk, 0, stream>>>(dst, deg, e);
  k_deg_rsqrt<<<gN, blk, 0, stream>>>(deg, n);

  // layer 1: relu(gcn(x, W1, b1))
  k_cvt_h   <<<gND, blk, 0, stream>>>(x, actH, nd);
  k_cvt_wT  <<<gW,  blk, 0, stream>>>(W1, wh);
  k_gemm128h<<<mTiles, blk, 0, stream>>>(actH, wh, hbuf, n);
  k_agg_init<<<gND, blk, 0, stream>>>(hbuf, deg, b1, act, n);
  k_scatter <<<gED, blk, 0, stream>>>(hbuf, deg, src, dst, act, e);
  k_relu_cvt<<<gND, blk, 0, stream>>>(act, actH, nd);

  // layer 2: relu(gcn(act, W2, b2))
  k_cvt_wT  <<<gW,  blk, 0, stream>>>(W2, wh);
  k_gemm128h<<<mTiles, blk, 0, stream>>>(actH, wh, hbuf, n);
  k_agg_init<<<gND, blk, 0, stream>>>(hbuf, deg, b2, act, n);
  k_scatter <<<gED, blk, 0, stream>>>(hbuf, deg, src, dst, act, e);
  k_relu_cvt<<<gND, blk, 0, stream>>>(act, actH, nd);

  // layer 3: gcn(act, W3, b3) (no relu)
  k_cvt_wT  <<<gW,  blk, 0, stream>>>(W3, wh);
  k_gemm128h<<<mTiles, blk, 0, stream>>>(actH, wh, hbuf, n);
  k_agg_init<<<gND, blk, 0, stream>>>(hbuf, deg, b3, act, n);
  k_scatter <<<gED, blk, 0, stream>>>(hbuf, deg, src, dst, act, e);

  // mean pool + head + log_softmax
  k_pool_zero <<<(NGR * DH + 255) / 256, blk, 0, stream>>>(pooled, counts);
  k_pool_accum<<<gND, blk, 0, stream>>>(act, bat, pooled, counts, n);
  k_head      <<<NGR, dim3(DH), 0, stream>>>(pooled, counts, Wl, bl, out);
}
